// VarianceAdaptor_69973607186998
// MI455X (gfx1250) — matrix operand order repaired and checked by reading, not compiled, not run on hardware
//
#include <hip/hip_runtime.h>
#include <hip/hip_bf16.h>

typedef __bf16 bf16;
typedef __attribute__((ext_vector_type(16))) __bf16 v16bf;
typedef __attribute__((ext_vector_type(8)))  __bf16 v8bf;
typedef __attribute__((ext_vector_type(8)))  float  v8f;
typedef __attribute__((ext_vector_type(4)))  unsigned int v4u;
typedef __attribute__((ext_vector_type(8)))  int v8i;
typedef __attribute__((ext_vector_type(4)))  int v4i;

#if __has_builtin(__builtin_amdgcn_tensor_load_to_lds) && __has_builtin(__builtin_amdgcn_s_wait_tensorcnt)
#define USE_TDM 1
#else
#define USE_TDM 0
#endif

static constexpr int BATCH  = 32;
static constexpr int TSRC   = 256;
static constexpr int DCH    = 256;
static constexpr int MAXMEL = 2048;
static constexpr int NBINS  = 256;

static constexpr size_t OUT_MAIN   = 0;
static constexpr size_t OUT_LOGDUR = (size_t)BATCH * MAXMEL * 512;
static constexpr size_t OUT_PITCH  = OUT_LOGDUR + (size_t)BATCH * TSRC;
static constexpr size_t OUT_MELLEN = OUT_PITCH + (size_t)BATCH * MAXMEL;

__device__ __forceinline__ float wave_sum(float v) {
    for (int m = 16; m >= 1; m >>= 1) v += __shfl_xor(v, m, 32);
    return v;
}

// Issue a 2D TDM load: nrows x 256 bf16 elements, row stride 256, into LDS.
__device__ __forceinline__ void tdm_load_2d(unsigned lds_byte_off, unsigned long long gaddr,
                                            int nrows) {
#if USE_TDM
    v4u g0;
    g0[0] = 1u;                                        // count=1, user desc
    g0[1] = lds_byte_off;                              // lds_addr
    g0[2] = (unsigned)(gaddr & 0xFFFFFFFFu);           // global_addr[31:0]
    g0[3] = (unsigned)((gaddr >> 32) & 0x1FFFFFFu) | 0x80000000u;  // addr[56:32] | type=2
    v8i g1;
    g1[0] = (int)(1u << 16);                           // data_size=1 (2 bytes/elem)
    g1[1] = (int)(256u << 16);                         // tensor_dim0 = 256 (bits 48..)
    g1[2] = (int)(((unsigned)nrows) << 16);            // tensor_dim1 = nrows (bits 80..)
    g1[3] = (int)(256u << 16);                         // tile_dim0 = 256 (bits 112..)
    g1[4] = (int)(unsigned)nrows;                      // tile_dim1 = nrows; tile_dim2 = 0
    g1[5] = 256;                                       // tensor_dim0_stride = 256
    g1[6] = 0;
    g1[7] = 0;
    v4i g2 = {0, 0, 0, 0};
    v4i g3 = {0, 0, 0, 0};
    v8i g4 = {0, 0, 0, 0, 0, 0, 0, 0};
    __builtin_amdgcn_tensor_load_to_lds(g0, g1, g2, g3, g4, 0);
    __builtin_amdgcn_s_wait_tensorcnt(0);
#else
    (void)lds_byte_off; (void)gaddr; (void)nrows;
#endif
}

// ---------------- prep kernels ----------------

__global__ void cvt_f32_to_bf16(const float* __restrict__ src, bf16* __restrict__ dst, int n) {
    int i = blockIdx.x * blockDim.x + threadIdx.x;
    if (i < n) dst[i] = (bf16)src[i];
}

__global__ void pack_conv_w(const float* __restrict__ w, bf16* __restrict__ wt, int n) {
    int i = blockIdx.x * blockDim.x + threadIdx.x;
    if (i >= n) return;
    int f = i / 768, rid = i % 768;
    int k = rid >> 8, d = rid & 255;
    wt[i] = (bf16)w[(f * 256 + d) * 3 + k];
}

__global__ void pack_proj_w(const float* __restrict__ w, bf16* __restrict__ wt, int n) {
    int i = blockIdx.x * blockDim.x + threadIdx.x;
    if (i >= n) return;
    int col = i >> 8, k = i & 255;
    wt[i] = (bf16)w[k * 512 + col];
}

// ---------------- length regulator indices ----------------
__global__ __launch_bounds__(256) void lr_index_kernel(const int* __restrict__ dur,
                                                       int* __restrict__ idx,
                                                       int* __restrict__ mel_len_i,
                                                       float* __restrict__ mel_len_out) {
    __shared__ int s[TSRC];
    int b = blockIdx.x, t = threadIdx.x;
    s[t] = dur[b * TSRC + t];
    __syncthreads();
    for (int off = 1; off < TSRC; off <<= 1) {
        int v = (t >= off) ? s[t - off] : 0;
        __syncthreads();
        if (t >= off) s[t] += v;
        __syncthreads();
    }
    if (t == 0) {
        int ml = min(s[TSRC - 1], MAXMEL);
        mel_len_i[b] = ml;
        mel_len_out[b] = (float)ml;
    }
    for (int r = 0; r < MAXMEL / TSRC; ++r) {
        int j = t + r * TSRC;
        int lo = 0, hi = TSRC;
        while (lo < hi) { int mid = (lo + hi) >> 1; if (s[mid] <= j) lo = mid + 1; else hi = mid; }
        idx[b * MAXMEL + j] = min(lo, TSRC - 1);
    }
}

__global__ __launch_bounds__(256) void gather_kernel(const float* __restrict__ lin,
                                                     const int* __restrict__ idx,
                                                     const int* __restrict__ mel_len,
                                                     float* __restrict__ xf, bf16* __restrict__ xh) {
    int row = blockIdx.x;
    int b = row >> 11, j = row & (MAXMEL - 1);
    int c = threadIdx.x;
    float v = 0.f;
    if (j < mel_len[b]) v = lin[((size_t)b * TSRC + idx[row]) * DCH + c];
    xf[(size_t)row * DCH + c] = v;
    xh[(size_t)row * DCH + c] = (bf16)v;
}

// ---------------- fused conv3(SAME)+bias+ReLU+LayerNorm, 32-row tile, TDM-staged A ----------------
__global__ __launch_bounds__(256) void conv_ln_kernel(const bf16* __restrict__ x,
                                                      const bf16* __restrict__ wt,   // [256][768]
                                                      const float* __restrict__ bias,
                                                      const float* __restrict__ gamma,
                                                      const float* __restrict__ beta,
                                                      bf16* __restrict__ out, int T) {
    __shared__ bf16  sA[34 * 256];
    __shared__ float sC[32][264];
    int tiles = T >> 5;
    int b  = blockIdx.x / tiles;
    int t0 = (blockIdx.x % tiles) << 5;
    int tid = threadIdx.x;
    const bf16* xb = x + (size_t)b * T * 256;

    // zero SAME-padding edge rows (LDS rows 0 / 33), disjoint from DMA target rows
    if (t0 == 0)      sA[tid] = (bf16)0.0f;
    if (t0 + 32 == T) sA[33 * 256 + tid] = (bf16)0.0f;

    int start = (t0 == 0) ? 0 : t0 - 1;
    int end   = min(t0 + 32, T - 1);
    int nrows = end - start + 1;
#if USE_TDM
    if (tid < 32) {
        unsigned lds_off = (unsigned)(size_t)(&sA[0]) + (unsigned)((start - (t0 - 1)) * 512);
        tdm_load_2d(lds_off, (unsigned long long)(size_t)(xb + (size_t)start * 256), nrows);
    }
#else
    for (int e = tid; e < nrows * 256; e += 256) {
        int r = e >> 8;
        sA[(start - (t0 - 1)) * 256 + e] = xb[(size_t)(start + r) * 256 + (e & 255)];
    }
#endif
    __syncthreads();

    int wave = tid >> 5, lane = tid & 31;
    int half = lane >> 4, l16 = lane & 15;
    int n0 = wave * 32, n1 = n0 + 16;
    v8f acc00 = {}, acc01 = {}, acc10 = {}, acc11 = {};
    const bf16* a0p = sA + l16 * 256;          // im2col rows 0..15
    const bf16* a1p = sA + (16 + l16) * 256;   // im2col rows 16..31
    const bf16* b0p = wt + (size_t)(n0 + l16) * 768;
    const bf16* b1p = wt + (size_t)(n1 + l16) * 768;

    for (int kk = 0; kk < 24; ++kk) {
        int kb = kk * 32;
        v16bf a0, a1, bb0, bb1;
        {
            v8bf lo0 = *(const v8bf*)(a0p + kb + half * 8);
            v8bf hi0 = *(const v8bf*)(a0p + kb + 16 + half * 8);
            v8bf lo1 = *(const v8bf*)(a1p + kb + half * 8);
            v8bf hi1 = *(const v8bf*)(a1p + kb + 16 + half * 8);
            for (int i = 0; i < 8; ++i) { a0[i] = lo0[i]; a0[8+i] = hi0[i]; a1[i] = lo1[i]; a1[8+i] = hi1[i]; }
            v8bf x0 = *(const v8bf*)(b0p + kb + half * 8);
            v8bf x1 = *(const v8bf*)(b0p + kb + 16 + half * 8);
            v8bf y0 = *(const v8bf*)(b1p + kb + half * 8);
            v8bf y1 = *(const v8bf*)(b1p + kb + 16 + half * 8);
            for (int i = 0; i < 8; ++i) { bb0[i] = x0[i]; bb0[8+i] = x1[i]; bb1[i] = y0[i]; bb1[8+i] = y1[i]; }
        }
        __builtin_prefetch(b0p + kb + 128, 0, 1);
        acc00 = __builtin_amdgcn_wmma_f32_16x16x32_bf16(false, a0, false, bb0, (short)0, acc00, false, false);
        acc01 = __builtin_amdgcn_wmma_f32_16x16x32_bf16(false, a0, false, bb1, (short)0, acc01, false, false);
        acc10 = __builtin_amdgcn_wmma_f32_16x16x32_bf16(false, a1, false, bb0, (short)0, acc10, false, false);
        acc11 = __builtin_amdgcn_wmma_f32_16x16x32_bf16(false, a1, false, bb1, (short)0, acc11, false, false);
    }

    int col0 = n0 + l16, col1 = n1 + l16;
    float bia0 = bias[col0], bia1 = bias[col1];
    for (int r = 0; r < 8; ++r) {
        int row = r + half * 8;
        sC[row][col0]      = fmaxf(acc00[r] + bia0, 0.f);
        sC[row][col1]      = fmaxf(acc01[r] + bia1, 0.f);
        sC[row + 16][col0] = fmaxf(acc10[r] + bia0, 0.f);
        sC[row + 16][col1] = fmaxf(acc11[r] + bia1, 0.f);
    }
    __syncthreads();

    for (int rr = 0; rr < 4; ++rr) {
        int row = wave * 4 + rr;
        float s = 0.f, s2 = 0.f, vals[8];
        for (int i = 0; i < 8; ++i) {
            float v = sC[row][lane + 32 * i];
            vals[i] = v; s += v; s2 += v * v;
        }
        s = wave_sum(s); s2 = wave_sum(s2);
        float mean = s * (1.f / 256.f);
        float var  = s2 * (1.f / 256.f) - mean * mean;
        float rs   = rsqrtf(var + 1e-5f);
        bf16* op = out + ((size_t)b * T + t0 + row) * 256;
        for (int i = 0; i < 8; ++i) {
            int c = lane + 32 * i;
            op[c] = (bf16)((vals[i] - mean) * rs * gamma[c] + beta[c]);
        }
    }
}

// ---------------- duration head ----------------
__global__ void rowdot_mask_kernel(const bf16* __restrict__ h, const float* __restrict__ lw,
                                   const float* __restrict__ lb, const unsigned char* __restrict__ mask,
                                   float* __restrict__ outp, int rows) {
    int gw = (blockIdx.x * blockDim.x + threadIdx.x) >> 5;
    int lane = threadIdx.x & 31;
    if (gw >= rows) return;
    const bf16* hp = h + (size_t)gw * 256;
    float s = 0.f;
    for (int i = 0; i < 8; ++i) { int c = lane + 32 * i; s += (float)hp[c] * lw[c]; }
    s = wave_sum(s);
    if (lane == 0) outp[gw] = mask[gw] ? 0.f : (s + lb[0]);
}

// ---------------- pitch head ----------------
__global__ void pitch_kernel(const bf16* __restrict__ h2, const float* __restrict__ lw,
                             const float* __restrict__ lb, const int* __restrict__ mel_len,
                             const float* __restrict__ bins, const float* __restrict__ emb,
                             const float* __restrict__ xf, float* __restrict__ pitch_out,
                             bf16* __restrict__ xa) {
    int row = (blockIdx.x * blockDim.x + threadIdx.x) >> 5;
    int lane = threadIdx.x & 31;
    if (row >= BATCH * MAXMEL) return;
    int b = row >> 11, j = row & (MAXMEL - 1);
    const bf16* hp = h2 + (size_t)row * 256;
    float s = 0.f;
    for (int i = 0; i < 8; ++i) { int c = lane + 32 * i; s += (float)hp[c] * lw[c]; }
    s = wave_sum(s);
    float p = (j < mel_len[b]) ? (s + lb[0]) : 0.f;
    if (lane == 0) pitch_out[row] = p;
    int lo = 0, hi = NBINS - 1;
    while (lo < hi) { int mid = (lo + hi) >> 1; if (bins[mid] < p) lo = mid + 1; else hi = mid; }
    const float* ep = emb + (size_t)lo * 256;
    const float* xp = xf + (size_t)row * 256;
    bf16* op = xa + (size_t)row * 256;
    for (int i = 0; i < 8; ++i) { int c = lane + 32 * i; op[c] = (bf16)(xp[c] + ep[c]); }
}

// ---------------- final projection GEMM, 32-row tile, TDM-staged A ----------------
__global__ __launch_bounds__(256) void proj_kernel(const bf16* __restrict__ xa,
                                                   const bf16* __restrict__ wt,   // [512][256]
                                                   const float* __restrict__ pb,
                                                   const int* __restrict__ mel_len,
                                                   float* __restrict__ outp) {
    __shared__ bf16  sA[32 * 256];
    __shared__ float sC[32][264];
    int row0 = blockIdx.x << 5;
    int ngrp = blockIdx.y;
    int tid = threadIdx.x;
#if USE_TDM
    if (tid < 32) {
        tdm_load_2d((unsigned)(size_t)(&sA[0]),
                    (unsigned long long)(size_t)(xa + (size_t)row0 * 256), 32);
    }
#else
    for (int e = tid; e < 32 * 256; e += 256) sA[e] = xa[(size_t)row0 * 256 + e];
#endif
    __syncthreads();

    int wave = tid >> 5, lane = tid & 31, half = lane >> 4, l16 = lane & 15;
    int c0 = wave * 32 + l16, c1 = c0 + 16;
    v8f acc00 = {}, acc01 = {}, acc10 = {}, acc11 = {};
    const bf16* a0p = sA + l16 * 256;
    const bf16* a1p = sA + (16 + l16) * 256;
    const bf16* b0p = wt + (size_t)(ngrp * 256 + c0) * 256;
    const bf16* b1p = wt + (size_t)(ngrp * 256 + c1) * 256;
    for (int kk = 0; kk < 8; ++kk) {
        int kb = kk * 32;
        v16bf a0, a1, bb0, bb1;
        {
            v8bf lo0 = *(const v8bf*)(a0p + kb + half * 8);
            v8bf hi0 = *(const v8bf*)(a0p + kb + 16 + half * 8);
            v8bf lo1 = *(const v8bf*)(a1p + kb + half * 8);
            v8bf hi1 = *(const v8bf*)(a1p + kb + 16 + half * 8);
            for (int i = 0; i < 8; ++i) { a0[i] = lo0[i]; a0[8+i] = hi0[i]; a1[i] = lo1[i]; a1[8+i] = hi1[i]; }
            v8bf x0 = *(const v8bf*)(b0p + kb + half * 8);
            v8bf x1 = *(const v8bf*)(b0p + kb + 16 + half * 8);
            v8bf y0 = *(const v8bf*)(b1p + kb + half * 8);
            v8bf y1 = *(const v8bf*)(b1p + kb + 16 + half * 8);
            for (int i = 0; i < 8; ++i) { bb0[i] = x0[i]; bb0[8+i] = x1[i]; bb1[i] = y0[i]; bb1[8+i] = y1[i]; }
        }
        acc00 = __builtin_amdgcn_wmma_f32_16x16x32_bf16(false, a0, false, bb0, (short)0, acc00, false, false);
        acc01 = __builtin_amdgcn_wmma_f32_16x16x32_bf16(false, a0, false, bb1, (short)0, acc01, false, false);
        acc10 = __builtin_amdgcn_wmma_f32_16x16x32_bf16(false, a1, false, bb0, (short)0, acc10, false, false);
        acc11 = __builtin_amdgcn_wmma_f32_16x16x32_bf16(false, a1, false, bb1, (short)0, acc11, false, false);
    }
    float bia0 = pb[ngrp * 256 + c0], bia1 = pb[ngrp * 256 + c1];
    for (int r = 0; r < 8; ++r) {
        int row = r + half * 8;
        sC[row][c0]      = acc00[r] + bia0;
        sC[row][c1]      = acc01[r] + bia1;
        sC[row + 16][c0] = acc10[r] + bia0;
        sC[row + 16][c1] = acc11[r] + bia1;
    }
    __syncthreads();
    for (int e = tid; e < 32 * 256; e += 256) {
        int r = e >> 8, c = e & 255;
        int row = row0 + r;
        int b = row >> 11, j = row & (MAXMEL - 1);
        float v = (j < mel_len[b]) ? sC[r][c] : 0.f;
        outp[(size_t)row * 512 + ngrp * 256 + c] = v;
    }
}

// ---------------- host launcher ----------------
extern "C" void kernel_launch(void* const* d_in, const int* in_sizes, int n_in,
                              void* d_out, int out_size, void* d_ws, size_t ws_size,
                              hipStream_t stream) {
    (void)in_sizes; (void)n_in; (void)out_size; (void)ws_size;
    const float* lin      = (const float*)d_in[0];
    const int*   duration = (const int*)d_in[1];
    const unsigned char* src_mask = (const unsigned char*)d_in[2];
    const float* dp_w1 = (const float*)d_in[3],  *dp_b1 = (const float*)d_in[4];
    const float* dp_g1 = (const float*)d_in[5],  *dp_be1 = (const float*)d_in[6];
    const float* dp_w2 = (const float*)d_in[7],  *dp_b2 = (const float*)d_in[8];
    const float* dp_g2 = (const float*)d_in[9],  *dp_be2 = (const float*)d_in[10];
    const float* dp_lw = (const float*)d_in[11], *dp_lb = (const float*)d_in[12];
    const float* pp_w1 = (const float*)d_in[13], *pp_b1 = (const float*)d_in[14];
    const float* pp_g1 = (const float*)d_in[15], *pp_be1 = (const float*)d_in[16];
    const float* pp_w2 = (const float*)d_in[17], *pp_b2 = (const float*)d_in[18];
    const float* pp_g2 = (const float*)d_in[19], *pp_be2 = (const float*)d_in[20];
    const float* pp_lw = (const float*)d_in[21], *pp_lb = (const float*)d_in[22];
    const float* bins  = (const float*)d_in[23];
    const float* emb   = (const float*)d_in[24];
    const float* proj_w = (const float*)d_in[25], *proj_b = (const float*)d_in[26];

    float* out = (float*)d_out;

    char* wsp = (char*)d_ws;
    size_t off = 0;
    auto take = [&](size_t bytes) { size_t r = off; off += (bytes + 255) & ~(size_t)255; return r; };
    const size_t LIN_N = (size_t)BATCH * TSRC * DCH;
    const size_t MEL_N = (size_t)BATCH * MAXMEL * DCH;
    bf16* lin_h  = (bf16*)(wsp + take(LIN_N * 2));
    bf16* wt_dp1 = (bf16*)(wsp + take(256 * 768 * 2));
    bf16* wt_dp2 = (bf16*)(wsp + take(256 * 768 * 2));
    bf16* wt_pp1 = (bf16*)(wsp + take(256 * 768 * 2));
    bf16* wt_pp2 = (bf16*)(wsp + take(256 * 768 * 2));
    bf16* wt_prj = (bf16*)(wsp + take(512 * 256 * 2));
    bf16* h1_dp  = (bf16*)(wsp + take(LIN_N * 2));
    bf16* h2_dp  = (bf16*)(wsp + take(LIN_N * 2));
    int*  idx    = (int*) (wsp + take((size_t)BATCH * MAXMEL * 4));
    int*  mel_i  = (int*) (wsp + take(BATCH * 4));
    float* x_f   = (float*)(wsp + take(MEL_N * 4));
    bf16* x_h    = (bf16*)(wsp + take(MEL_N * 2));
    bf16* p1     = (bf16*)(wsp + take(MEL_N * 2));
    bf16* p2     = (bf16*)(wsp + take(MEL_N * 2));
    bf16* xa     = (bf16*)(wsp + take(MEL_N * 2));

    cvt_f32_to_bf16<<<(LIN_N + 255) / 256, 256, 0, stream>>>(lin, lin_h, (int)LIN_N);
    pack_conv_w<<<(196608 + 255) / 256, 256, 0, stream>>>(dp_w1, wt_dp1, 196608);
    pack_conv_w<<<(196608 + 255) / 256, 256, 0, stream>>>(dp_w2, wt_dp2, 196608);
    pack_conv_w<<<(196608 + 255) / 256, 256, 0, stream>>>(pp_w1, wt_pp1, 196608);
    pack_conv_w<<<(196608 + 255) / 256, 256, 0, stream>>>(pp_w2, wt_pp2, 196608);
    pack_proj_w<<<(131072 + 255) / 256, 256, 0, stream>>>(proj_w, wt_prj, 131072);

    lr_index_kernel<<<BATCH, 256, 0, stream>>>(duration, idx, mel_i, out + OUT_MELLEN);

    conv_ln_kernel<<<BATCH * (TSRC / 32), 256, 0, stream>>>(lin_h, wt_dp1, dp_b1, dp_g1, dp_be1, h1_dp, TSRC);
    conv_ln_kernel<<<BATCH * (TSRC / 32), 256, 0, stream>>>(h1_dp, wt_dp2, dp_b2, dp_g2, dp_be2, h2_dp, TSRC);
    rowdot_mask_kernel<<<(BATCH * TSRC) / 8, 256, 0, stream>>>(h2_dp, dp_lw, dp_lb, src_mask,
                                                               out + OUT_LOGDUR, BATCH * TSRC);

    gather_kernel<<<BATCH * MAXMEL, 256, 0, stream>>>(lin, idx, mel_i, x_f, x_h);

    conv_ln_kernel<<<BATCH * (MAXMEL / 32), 256, 0, stream>>>(x_h, wt_pp1, pp_b1, pp_g1, pp_be1, p1, MAXMEL);
    conv_ln_kernel<<<BATCH * (MAXMEL / 32), 256, 0, stream>>>(p1, wt_pp2, pp_b2, pp_g2, pp_be2, p2, MAXMEL);
    pitch_kernel<<<(BATCH * MAXMEL) / 8, 256, 0, stream>>>(p2, pp_lw, pp_lb, mel_i, bins, emb, x_f,
                                                           out + OUT_PITCH, xa);

    dim3 pg((BATCH * MAXMEL) / 32, 2);
    proj_kernel<<<pg, 256, 0, stream>>>(xa, wt_prj, proj_b, mel_i, out + OUT_MAIN);
}
